// PaliGemmaWithExpertModel_55903294325286
// MI455X (gfx1250) — matrix operand order repaired and checked by reading, not compiled, not run on hardware
//
#include <hip/hip_runtime.h>
#include <math.h>

typedef __bf16 bf16_t;
typedef __attribute__((ext_vector_type(16))) __bf16 v16bf;
typedef __attribute__((ext_vector_type(8)))  __bf16 v8bf;
typedef __attribute__((ext_vector_type(8)))  float  v8f;

// ---------------------------------------------------------------------------
// Device helpers: WMMA fragment loaders (wave32), contiguous k = 16*half + i in A and B alike
// ---------------------------------------------------------------------------
__device__ __forceinline__ v8f wmma_bf16f32(v16bf a, v16bf b, v8f c) {
  return __builtin_amdgcn_wmma_f32_16x16x32_bf16(false, a, false, b, (short)0, c,
                                                 false, false);
}

// A-frag 16x32 from row-major bf16 (lda elements). lanes 0-15/16-31 both hold
// rows 0-15; elem i -> K = half*16 + i.
__device__ __forceinline__ v16bf load_a_frag(const bf16_t* A, int lda, int row0, int k0) {
  int lane = threadIdx.x & 31;
  int hf = lane >> 4;
  const bf16_t* p = A + (size_t)(row0 + (lane & 15)) * lda + k0 + hf * 16;
  v8bf lo = *(const v8bf*)p;        // K = half*16 + 0..7
  v8bf hi = *(const v8bf*)(p + 8);  // K = half*16 + 8..15
  v16bf a;
#pragma unroll
  for (int i = 0; i < 8; ++i) { a[i] = lo[i]; a[i + 8] = hi[i]; }
  return a;
}

// B-frag 32x16 from pre-swizzled weights: lane-major, 16 contiguous bf16/lane.
__device__ __forceinline__ v16bf load_wp_frag(const bf16_t* fp) {
  v8bf lo = *(const v8bf*)fp;
  v8bf hi = *(const v8bf*)(fp + 8);
  v16bf b;
#pragma unroll
  for (int i = 0; i < 8; ++i) { b[i] = lo[i]; b[i + 8] = hi[i]; }
  return b;
}

// B-frag for Q*K^T: B[k=d][col=key] = K[key0+col][k0 + khalf*16 + i]
// K stored row-major [S][hd] -> per-lane contiguous in d.
__device__ __forceinline__ v16bf load_kT_frag(const bf16_t* K, int ldk, int key0, int k0) {
  int lane = threadIdx.x & 31;
  const bf16_t* p = K + (size_t)(key0 + (lane & 15)) * ldk + k0 + (lane >> 4) * 16;
  v8bf lo = *(const v8bf*)p;
  v8bf hi = *(const v8bf*)(p + 8);
  v16bf b;
#pragma unroll
  for (int i = 0; i < 8; ++i) { b[i] = lo[i]; b[i + 8] = hi[i]; }
  return b;
}

// B-frag for P*V: V stored transposed [hd][S] -> contiguous in key per lane.
__device__ __forceinline__ v16bf load_vt_frag(const bf16_t* Vt, int S, int key0, int d0) {
  int lane = threadIdx.x & 31;
  const bf16_t* p = Vt + (size_t)(d0 + (lane & 15)) * S + key0 + (lane >> 4) * 16;
  v8bf lo = *(const v8bf*)p;
  v8bf hi = *(const v8bf*)(p + 8);
  v16bf b;
#pragma unroll
  for (int i = 0; i < 8; ++i) { b[i] = lo[i]; b[i + 8] = hi[i]; }
  return b;
}

// A-frag from LDS P tile (16 rows x 32 keys, row-major)
__device__ __forceinline__ v16bf load_p_frag(const bf16_t* p) {
  int lane = threadIdx.x & 31;
  int row = lane & 15, hf = lane >> 4;
  v16bf a;
#pragma unroll
  for (int i = 0; i < 8; ++i) a[i] = p[row * 32 + hf * 16 + i];
#pragma unroll
  for (int i = 0; i < 8; ++i) a[i + 8] = p[row * 32 + hf * 16 + 8 + i];
  return a;
}

// ---------------------------------------------------------------------------
// Weight packing: f32 [K][N] -> bf16 B-fragments [ntile][ktile][lane][16]
// ---------------------------------------------------------------------------
__global__ __launch_bounds__(32) void pack_w_kernel(const float* __restrict__ W,
                                                    bf16_t* __restrict__ Wp,
                                                    int K, int N) {
  int kt = blockIdx.x, nt = blockIdx.y;
  int lane = threadIdx.x;
  int col = nt * 16 + (lane & 15);
  int kbase = kt * 32 + (lane >> 4) * 16;
  int ktn = K >> 5;
  bf16_t* dst = Wp + ((size_t)nt * ktn + kt) * 512 + lane * 16;
#pragma unroll
  for (int i = 0; i < 16; ++i) dst[i] = (bf16_t)W[(size_t)(kbase + i) * N + col];
}

// ---------------------------------------------------------------------------
// GEMM: C[M][N] (+= if resid, + bias) = A_bf16[M][K] @ Wp(bf16 frags)
// block = 128 threads (4 waves); each wave: 2 row-tiles x 4 col-tiles.
// Software-pipelined: fragments for k+1 are loaded while k's WMMAs execute,
// so the scheduler can wait on loadcnt > 0 instead of draining to zero.
// ---------------------------------------------------------------------------
__global__ __launch_bounds__(128) void gemm_bf16_kernel(
    const bf16_t* __restrict__ A, const bf16_t* __restrict__ Wp,
    const float* __restrict__ bias, float* __restrict__ C,
    int M, int N, int K, int resid) {
  int lane = threadIdx.x & 31;
  int wave = threadIdx.x >> 5;
  int row0 = blockIdx.x * 32;  // two 16-row tiles
  int ntiles = N >> 4;
  int nt0 = blockIdx.y * 16 + wave * 4;
  int ktn = K >> 5;
  int ntc[4];
#pragma unroll
  for (int j = 0; j < 4; ++j) ntc[j] = (nt0 + j < ntiles) ? (nt0 + j) : (ntiles - 1);

  v8f acc0[4] = {}, acc1[4] = {};
  // prologue: fragments for kt = 0
  v16bf a0 = load_a_frag(A, K, row0, 0);
  v16bf a1 = load_a_frag(A, K, row0 + 16, 0);
  v16bf bb[4];
#pragma unroll
  for (int j = 0; j < 4; ++j)
    bb[j] = load_wp_frag(Wp + (size_t)ntc[j] * ktn * 512 + lane * 16);

  for (int kt = 0; kt < ktn; ++kt) {
    int ktn1 = (kt + 1 < ktn) ? (kt + 1) : kt;  // clamp: last iter reloads (uniform)
    // prefetch the k+2 weight fragments toward near caches
    int ktn2 = (kt + 2 < ktn) ? (kt + 2) : kt;
    __builtin_prefetch(Wp + ((size_t)ntc[0] * ktn + ktn2) * 512 + lane * 16, 0, 3);
    __builtin_prefetch(Wp + ((size_t)ntc[2] * ktn + ktn2) * 512 + lane * 16, 0, 3);
    // issue next-iteration loads
    v16bf a0n = load_a_frag(A, K, row0, ktn1 * 32);
    v16bf a1n = load_a_frag(A, K, row0 + 16, ktn1 * 32);
    v16bf bbn[4];
#pragma unroll
    for (int j = 0; j < 4; ++j)
      bbn[j] = load_wp_frag(Wp + ((size_t)ntc[j] * ktn + ktn1) * 512 + lane * 16);
    // compute on current fragments (8 independent WMMAs)
#pragma unroll
    for (int j = 0; j < 4; ++j) {
      acc0[j] = wmma_bf16f32(a0, bb[j], acc0[j]);
      acc1[j] = wmma_bf16f32(a1, bb[j], acc1[j]);
    }
    a0 = a0n; a1 = a1n;
#pragma unroll
    for (int j = 0; j < 4; ++j) bb[j] = bbn[j];
  }

  int col = lane & 15;
  int rb0 = row0 + (lane >> 4) * 8;
#pragma unroll
  for (int j = 0; j < 4; ++j) {
    if (nt0 + j >= ntiles) continue;
    int c0 = (nt0 + j) * 16 + col;
    float bsv = bias ? bias[c0] : 0.0f;
#pragma unroll
    for (int i = 0; i < 8; ++i) {
      size_t idx0 = (size_t)(rb0 + i) * N + c0;
      size_t idx1 = (size_t)(rb0 + 16 + i) * N + c0;
      float v0 = acc0[j][i] + bsv;
      float v1 = acc1[j][i] + bsv;
      if (resid) { v0 += C[idx0]; v1 += C[idx1]; }
      C[idx0] = v0;
      C[idx1] = v1;
    }
  }
}

// ---------------------------------------------------------------------------
// Fused flash attention: one wave per (q-tile16, head, batch), kv=1.
// Q: [B][H][S][HD] bf16 (rope applied); K: [B][S][HD] bf16; Vt: [B][HD][S] bf16
// O: [B*S][H*HD] bf16 (GEMM-ready activations for the wo projection)
// ---------------------------------------------------------------------------
template <int HD>
__global__ __launch_bounds__(32) void flash_attn_kernel(
    const bf16_t* __restrict__ Q, const bf16_t* __restrict__ Km,
    const bf16_t* __restrict__ Vt, bf16_t* __restrict__ O,
    int S, int H, float scale) {
  constexpr int KT = HD / 32;  // A-frags along d for Q*K^T
  constexpr int NT = HD / 16;  // output column tiles for P*V
  int lane = threadIdx.x & 31;
  int qt = blockIdx.x, h = blockIdx.y, b = blockIdx.z;
  const bf16_t* Qh = Q + (size_t)(b * H + h) * S * HD;
  const bf16_t* Kb = Km + (size_t)b * S * HD;
  const bf16_t* Vb = Vt + (size_t)b * HD * S;
  __shared__ bf16_t pbuf[16 * 32];

  v16bf aq[KT];
#pragma unroll
  for (int t = 0; t < KT; ++t) aq[t] = load_a_frag(Qh, HD, qt * 16, t * 32);

  v8f acc[NT] = {};
  float m[8], l[8];
#pragma unroll
  for (int i = 0; i < 8; ++i) { m[i] = -3.0e38f; l[i] = 0.0f; }

  int col = lane & 15, half = lane >> 4;
  int qrow_base = qt * 16 + half * 8;
  int nblk = (qt * 16 + 15) / 32 + 1;  // causal: key blocks of 32 covering keys <= q

  for (int kb = 0; kb < nblk; ++kb) {
    int kpos = kb * 32;
    // load all K fragments up front, then run the WMMA chain
    v16bf bk0[KT], bk1[KT];
#pragma unroll
    for (int t = 0; t < KT; ++t) {
      bk0[t] = load_kT_frag(Kb, HD, kpos, t * 32);
      bk1[t] = load_kT_frag(Kb, HD, kpos + 16, t * 32);
    }
    v8f s0 = {}, s1 = {};
#pragma unroll
    for (int t = 0; t < KT; ++t) {
      s0 = wmma_bf16f32(aq[t], bk0[t], s0);
      s1 = wmma_bf16f32(aq[t], bk1[t], s1);
    }
    // V fragments don't depend on P: issue them now so they overlap the
    // softmax VALU work below.
    v16bf vf[NT];
#pragma unroll
    for (int n = 0; n < NT; ++n) vf[n] = load_vt_frag(Vb, S, kpos, n * 16);

    float pr0[8], pr1[8], corr[8];
#pragma unroll
    for (int i = 0; i < 8; ++i) {
      int qr = qrow_base + i;
      float v0 = s0[i] * scale, v1 = s1[i] * scale;
      if (kpos + col > qr) v0 = -1.0e30f;         // causal mask (lane select)
      if (kpos + 16 + col > qr) v1 = -1.0e30f;
      float mx = fmaxf(v0, v1);
      for (int off = 1; off < 16; off <<= 1) mx = fmaxf(mx, __shfl_xor(mx, off, 32));
      float mn = fmaxf(m[i], mx);
      float c = __expf(m[i] - mn);
      float p0 = __expf(v0 - mn);
      float p1 = __expf(v1 - mn);
      float sum = p0 + p1;
      for (int off = 1; off < 16; off <<= 1) sum += __shfl_xor(sum, off, 32);
      l[i] = l[i] * c + sum;
      m[i] = mn;
      corr[i] = c;
      pr0[i] = p0; pr1[i] = p1;
    }
#pragma unroll
    for (int n = 0; n < NT; ++n)
#pragma unroll
      for (int i = 0; i < 8; ++i) acc[n][i] *= corr[i];
    // stage P (16x32) through LDS to re-shape C-layout -> A-fragment
#pragma unroll
    for (int i = 0; i < 8; ++i) {
      int r = half * 8 + i;
      pbuf[r * 32 + col]      = (bf16_t)pr0[i];
      pbuf[r * 32 + 16 + col] = (bf16_t)pr1[i];
    }
    __syncthreads();
    v16bf ap = load_p_frag(pbuf);
#pragma unroll
    for (int n = 0; n < NT; ++n) acc[n] = wmma_bf16f32(ap, vf[n], acc[n]);
    __syncthreads();
  }

  float linv[8];
#pragma unroll
  for (int i = 0; i < 8; ++i) linv[i] = 1.0f / l[i];
#pragma unroll
  for (int n = 0; n < NT; ++n)
#pragma unroll
    for (int i = 0; i < 8; ++i) {
      int qr = qrow_base + i;
      O[(size_t)(b * S + qr) * (H * HD) + h * HD + n * 16 + col] =
          (bf16_t)(acc[n][i] * linv[i]);
    }
}

// ---------------------------------------------------------------------------
// Elementwise / norm kernels
// ---------------------------------------------------------------------------
__global__ void scale_copy_kernel(const float* __restrict__ src, float* __restrict__ dst,
                                  float scale, int n) {
  int i = blockIdx.x * 256 + threadIdx.x;
  if (i < n) dst[i] = src[i] * scale;
}

__global__ __launch_bounds__(256) void rmsnorm_bf16_kernel(
    const float* __restrict__ h, const float* __restrict__ w,
    bf16_t* __restrict__ out, int D) {
  int row = blockIdx.x;
  const float* hr = h + (size_t)row * D;
  __shared__ float red[256];
  float ss = 0.0f;
  for (int d = threadIdx.x; d < D; d += 256) { float v = hr[d]; ss += v * v; }
  red[threadIdx.x] = ss;
  __syncthreads();
  for (int s = 128; s > 0; s >>= 1) {
    if (threadIdx.x < s) red[threadIdx.x] += red[threadIdx.x + s];
    __syncthreads();
  }
  float inv = rsqrtf(red[0] / (float)D + 1e-6f);
  for (int d = threadIdx.x; d < D; d += 256)
    out[(size_t)row * D + d] = (bf16_t)(hr[d] * inv * (1.0f + w[d]));
}

// final expert norm + suffix slice -> f32 output
__global__ __launch_bounds__(256) void rmsnorm_slice_kernel(
    const float* __restrict__ h, const float* __restrict__ w,
    float* __restrict__ out, int D, int Srow, int row_off, int SS) {
  int r = blockIdx.x % SS, b = blockIdx.x / SS;
  const float* hr = h + (size_t)(b * Srow + row_off + r) * D;
  __shared__ float red[256];
  float ss = 0.0f;
  for (int d = threadIdx.x; d < D; d += 256) { float v = hr[d]; ss += v * v; }
  red[threadIdx.x] = ss;
  __syncthreads();
  for (int s = 128; s > 0; s >>= 1) {
    if (threadIdx.x < s) red[threadIdx.x] += red[threadIdx.x + s];
    __syncthreads();
  }
  float inv = rsqrtf(red[0] / (float)D + 1e-6f);
  for (int d = threadIdx.x; d < D; d += 256)
    out[(size_t)blockIdx.x * D + d] = hr[d] * inv * (1.0f + w[d]);
}

// RoPE: f32 [B*S][H*hd] -> bf16 [B][H][S][hd]  (GPT-NeoX half-split rotation)
__global__ void rope_kernel(const float* __restrict__ qf, bf16_t* __restrict__ out,
                            int B_, int S, int H, int hd) {
  int h2 = hd >> 1;
  int idx = blockIdx.x * 256 + threadIdx.x;
  int total = B_ * S * H * h2;
  if (idx >= total) return;
  int j = idx % h2; int t = idx / h2;
  int hh = t % H; t /= H;
  int s = t % S; int b = t / S;
  float inv = __expf(-((2.0f * j) / (float)hd) * 9.210340371976184f);  // ln(10000)
  float ang = (float)s * inv;
  float c = cosf(ang), sn = sinf(ang);
  const float* base = qf + ((size_t)(b * S + s) * H + hh) * hd;
  float x1 = base[j], x2 = base[j + h2];
  bf16_t* ob = out + ((size_t)(b * H + hh) * S + s) * hd;
  ob[j]      = (bf16_t)(x1 * c - x2 * sn);
  ob[j + h2] = (bf16_t)(x2 * c + x1 * sn);
}

// V transpose: f32 [B*S][hd] -> bf16 [B][hd][S]
__global__ void transpose_v_kernel(const float* __restrict__ vf, bf16_t* __restrict__ vt,
                                   int B_, int S, int hd) {
  int idx = blockIdx.x * 256 + threadIdx.x;
  int total = B_ * S * hd;
  if (idx >= total) return;
  int d = idx % hd; int t = idx / hd;
  int s = t % S; int b = t / S;
  vt[((size_t)(b * hd + d)) * S + s] = (bf16_t)vf[idx];
}

__global__ void gelu_mul_kernel(const float* __restrict__ g, const float* __restrict__ u,
                                bf16_t* __restrict__ a, int n) {
  int i = blockIdx.x * 256 + threadIdx.x;
  if (i < n) {
    float x = g[i];
    float t = tanhf(0.7978845608028654f * (x + 0.044715f * x * x * x));
    a[i] = (bf16_t)(0.5f * x * (1.0f + t) * u[i]);
  }
}

// combined = [proj(prefix) ; suffix] * sqrt(Dexp)
__global__ void combine_kernel(const float* __restrict__ proj, const float* __restrict__ suf,
                               float* __restrict__ h, float scale) {
  const int D = 512, Srow = 816, SP = 768, SS = 48;
  int idx = blockIdx.x * 256 + threadIdx.x;
  int total = 4 * Srow * D;
  if (idx >= total) return;
  int d = idx % D; int t = idx / D;
  int s = t % Srow; int b = t / Srow;
  float v = (s < SP) ? proj[(size_t)(b * SP + s) * D + d]
                     : suf[(size_t)(b * SS + (s - SP)) * D + d];
  h[idx] = v * scale;
}

// ---------------------------------------------------------------------------
// Host orchestration
// ---------------------------------------------------------------------------
static inline int cdiv(int a, int b) { return (a + b - 1) / b; }

static void run_gemma_layers(hipStream_t stream, float* h, int Bn, int S, int D,
                             int Mlp, int H, int hd, int L,
                             const float* wq, const float* wk, const float* wv,
                             const float* wo, const float* ln1, const float* ln2,
                             const float* wg, const float* wu, const float* wd,
                             bf16_t* x_bf, float* q_f, bf16_t* qr_bf, float* k_f,
                             bf16_t* kr_bf, float* v_f, bf16_t* vt_bf, bf16_t* o_bf,
                             float* g_f, float* u_f, bf16_t* a_bf, bf16_t* wp) {
  int M = Bn * S;
  int HHD = H * hd;
  float att_scale = rsqrtf((float)hd);
  for (int l = 0; l < L; ++l) {
    const float* wql = wq + (size_t)l * D * HHD;
    const float* wkl = wk + (size_t)l * D * hd;
    const float* wvl = wv + (size_t)l * D * hd;
    const float* wol = wo + (size_t)l * HHD * D;
    const float* wgl = wg + (size_t)l * D * Mlp;
    const float* wul = wu + (size_t)l * D * Mlp;
    const float* wdl = wd + (size_t)l * Mlp * D;

    // x = rmsnorm(h, ln1)
    rmsnorm_bf16_kernel<<<M, 256, 0, stream>>>(h, ln1 + (size_t)l * D, x_bf, D);
    // q, k, v projections
    pack_w_kernel<<<dim3(D / 32, HHD / 16), 32, 0, stream>>>(wql, wp, D, HHD);
    gemm_bf16_kernel<<<dim3(M / 32, cdiv(HHD, 256)), 128, 0, stream>>>(
        x_bf, wp, nullptr, q_f, M, HHD, D, 0);
    pack_w_kernel<<<dim3(D / 32, hd / 16), 32, 0, stream>>>(wkl, wp, D, hd);
    gemm_bf16_kernel<<<dim3(M / 32, cdiv(hd, 256)), 128, 0, stream>>>(
        x_bf, wp, nullptr, k_f, M, hd, D, 0);
    pack_w_kernel<<<dim3(D / 32, hd / 16), 32, 0, stream>>>(wvl, wp, D, hd);
    gemm_bf16_kernel<<<dim3(M / 32, cdiv(hd, 256)), 128, 0, stream>>>(
        x_bf, wp, nullptr, v_f, M, hd, D, 0);
    // rope(q), rope(k), transpose(v)
    rope_kernel<<<cdiv(Bn * S * H * (hd / 2), 256), 256, 0, stream>>>(q_f, qr_bf, Bn, S, H, hd);
    rope_kernel<<<cdiv(Bn * S * 1 * (hd / 2), 256), 256, 0, stream>>>(k_f, kr_bf, Bn, S, 1, hd);
    transpose_v_kernel<<<cdiv(Bn * S * hd, 256), 256, 0, stream>>>(v_f, vt_bf, Bn, S, hd);
    // fused flash attention
    if (hd == 128)
      flash_attn_kernel<128><<<dim3(S / 16, H, Bn), 32, 0, stream>>>(
          qr_bf, kr_bf, vt_bf, o_bf, S, H, att_scale);
    else
      flash_attn_kernel<64><<<dim3(S / 16, H, Bn), 32, 0, stream>>>(
          qr_bf, kr_bf, vt_bf, o_bf, S, H, att_scale);
    // h += o @ wo
    pack_w_kernel<<<dim3(HHD / 32, D / 16), 32, 0, stream>>>(wol, wp, HHD, D);
    gemm_bf16_kernel<<<dim3(M / 32, cdiv(D, 256)), 128, 0, stream>>>(
        o_bf, wp, nullptr, h, M, D, HHD, 1);
    // MLP
    rmsnorm_bf16_kernel<<<M, 256, 0, stream>>>(h, ln2 + (size_t)l * D, x_bf, D);
    pack_w_kernel<<<dim3(D / 32, Mlp / 16), 32, 0, stream>>>(wgl, wp, D, Mlp);
    gemm_bf16_kernel<<<dim3(M / 32, cdiv(Mlp, 256)), 128, 0, stream>>>(
        x_bf, wp, nullptr, g_f, M, Mlp, D, 0);
    pack_w_kernel<<<dim3(D / 32, Mlp / 16), 32, 0, stream>>>(wul, wp, D, Mlp);
    gemm_bf16_kernel<<<dim3(M / 32, cdiv(Mlp, 256)), 128, 0, stream>>>(
        x_bf, wp, nullptr, u_f, M, Mlp, D, 0);
    gelu_mul_kernel<<<cdiv(M * Mlp, 256), 256, 0, stream>>>(g_f, u_f, a_bf, M * Mlp);
    pack_w_kernel<<<dim3(Mlp / 32, D / 16), 32, 0, stream>>>(wdl, wp, Mlp, D);
    gemm_bf16_kernel<<<dim3(M / 32, cdiv(D, 256)), 128, 0, stream>>>(
        a_bf, wp, nullptr, h, M, D, Mlp, 1);
  }
}

extern "C" void kernel_launch(void* const* d_in, const int* in_sizes, int n_in,
                              void* d_out, int out_size, void* d_ws, size_t ws_size,
                              hipStream_t stream) {
  (void)in_sizes; (void)n_in; (void)out_size; (void)ws_size;
  const float* prefix   = (const float*)d_in[0];
  const float* suffix   = (const float*)d_in[1];
  const float* vlm_wq   = (const float*)d_in[2];
  const float* vlm_wk   = (const float*)d_in[3];
  const float* vlm_wv   = (const float*)d_in[4];
  const float* vlm_wo   = (const float*)d_in[5];
  const float* vlm_ln1  = (const float*)d_in[6];
  const float* vlm_ln2  = (const float*)d_in[7];
  const float* vlm_wg   = (const float*)d_in[8];
  const float* vlm_wu   = (const float*)d_in[9];
  const float* vlm_wd   = (const float*)d_in[10];
  const float* vlm_fn   = (const float*)d_in[11];
  const float* exp_wq   = (const float*)d_in[12];
  const float* exp_wk   = (const float*)d_in[13];
  const float* exp_wv   = (const float*)d_in[14];
  const float* exp_wo   = (const float*)d_in[15];
  const float* exp_ln1  = (const float*)d_in[16];
  const float* exp_ln2  = (const float*)d_in[17];
  const float* exp_wg   = (const float*)d_in[18];
  const float* exp_wu   = (const float*)d_in[19];
  const float* exp_wd   = (const float*)d_in[20];
  const float* exp_fn   = (const float*)d_in[21];
  const float* proj_w   = (const float*)d_in[22];
  const float* proj_b   = (const float*)d_in[23];
  float* out = (float*)d_out;

  // workspace carve (bump allocator, 256B aligned); sized for the VLM maxima
  char* ws = (char*)d_ws;
  auto alloc = [&](size_t bytes) -> void* {
    void* p = (void*)ws;
    ws += (bytes + 255) & ~(size_t)255;
    return p;
  };
  float*  h_vlm  = (float*)alloc((size_t)3072 * 1024 * 4);
  float*  h_exp  = (float*)alloc((size_t)3264 * 512 * 4);
  bf16_t* x_bf   = (bf16_t*)alloc((size_t)3072 * 1024 * 2);
  float*  q_f    = (float*)alloc((size_t)3072 * 1024 * 4);
  bf16_t* qr_bf  = (bf16_t*)alloc((size_t)3072 * 1024 * 2);
  float*  k_f    = (float*)alloc((size_t)3072 * 128 * 4);
  bf16_t* kr_bf  = (bf16_t*)alloc((size_t)3072 * 128 * 2);
  float*  v_f    = (float*)alloc((size_t)3072 * 128 * 4);
  bf16_t* vt_bf  = (bf16_t*)alloc((size_t)3072 * 128 * 2);
  bf16_t* o_bf   = (bf16_t*)alloc((size_t)3072 * 1024 * 2);
  float*  g_f    = (float*)alloc((size_t)3072 * 4096 * 4);
  float*  u_f    = (float*)alloc((size_t)3072 * 4096 * 4);
  bf16_t* a_bf   = (bf16_t*)alloc((size_t)3072 * 4096 * 2);
  bf16_t* wp     = (bf16_t*)alloc((size_t)1024 * 4096 * 2);
  float*  proj_f = (float*)alloc((size_t)3072 * 512 * 4);

  // ---- VLM tower: h = prefix * sqrt(1024), 4 layers ----
  scale_copy_kernel<<<cdiv(4 * 768 * 1024, 256), 256, 0, stream>>>(
      prefix, h_vlm, 32.0f, 4 * 768 * 1024);
  run_gemma_layers(stream, h_vlm, 4, 768, 1024, 4096, 8, 128, 4,
                   vlm_wq, vlm_wk, vlm_wv, vlm_wo, vlm_ln1, vlm_ln2,
                   vlm_wg, vlm_wu, vlm_wd,
                   x_bf, q_f, qr_bf, k_f, kr_bf, v_f, vt_bf, o_bf, g_f, u_f, a_bf, wp);
  // final VLM norm (bf16 for the projection GEMM)
  rmsnorm_bf16_kernel<<<3072, 256, 0, stream>>>(h_vlm, vlm_fn, x_bf, 1024);
  // prefix_proj = normed @ proj_w + proj_b
  pack_w_kernel<<<dim3(1024 / 32, 512 / 16), 32, 0, stream>>>(proj_w, wp, 1024, 512);
  gemm_bf16_kernel<<<dim3(3072 / 32, cdiv(512, 256)), 128, 0, stream>>>(
      x_bf, wp, proj_b, proj_f, 3072, 512, 1024, 0);
  // combined = [proj ; suffix] * sqrt(512)
  combine_kernel<<<cdiv(4 * 816 * 512, 256), 256, 0, stream>>>(
      proj_f, suffix, h_exp, 22.627416997969522f);

  // ---- Expert tower: S=816, D=512, hd=64 ----
  run_gemma_layers(stream, h_exp, 4, 816, 512, 2048, 8, 64, 4,
                   exp_wq, exp_wk, exp_wv, exp_wo, exp_ln1, exp_ln2,
                   exp_wg, exp_wu, exp_wd,
                   x_bf, q_f, qr_bf, k_f, kr_bf, v_f, vt_bf, o_bf, g_f, u_f, a_bf, wp);

  // final norm + suffix slice -> d_out (f32, [4][48][512])
  rmsnorm_slice_kernel<<<4 * 48, 256, 0, stream>>>(h_exp, exp_fn, out, 512, 816, 768, 48);
}
